// PGExplainer_66571993088771
// MI455X (gfx1250) — matrix operand order repaired and checked by reading, not compiled, not run on hardware
//
#include <hip/hip_runtime.h>

// ---------------------------------------------------------------------------
// PGExplainer edge-MLP for MI455X (gfx1250, wave32, WMMA).
//
// Roofline: ~512 MB of gathered embeddings dominates (Z = 25.6 MB, fully
// L2-resident at 192 MB); 19 GFLOP of matmul needs >870 TF/s to stay
// memory-bound -> v_wmma_f32_16x16x32_f16 (f32 accumulate).
// Per 16-edge tile: 16 b128 gathers, 20 WMMAs, LDS f16 transpose, shuffle
// reduce. Edge-index loads + Z-row prefetch are software-pipelined one tile
// ahead to break the idx->gather dependent-latency chain.
// ---------------------------------------------------------------------------

typedef __attribute__((ext_vector_type(16))) _Float16 v16h;
typedef __attribute__((ext_vector_type(8)))  _Float16 v8h;
typedef __attribute__((ext_vector_type(8)))  float    v8f;
typedef __attribute__((ext_vector_type(4)))  float    v4f;

#define DD   64
#define HH1  64
#define HH2  20
#define WPB  8            // waves per block (256 threads)
#define LSTR 72           // LDS row stride in halfs (64 + 8 pad -> conflict-free b128)

// ---- tiny setup kernel: c1[j] = b1[j] + sum_k Z[node,k] * W1[2D+k, j] ----
__global__ void pg_precompute_c1(const float* __restrict__ Z,
                                 const int*   __restrict__ node_idx_p,
                                 const float* __restrict__ W1,
                                 const float* __restrict__ b1,
                                 float* __restrict__ c1) {
  int j = threadIdx.x;                       // 0..63
  const float* zv = Z + (size_t)node_idx_p[0] * DD;
  float acc = b1[j];
  for (int k = 0; k < DD; ++k)
    acc += zv[k] * W1[(2 * DD + k) * HH1 + j];
  c1[j] = acc;
}

// ---- main kernel: 1 wave == 1 tile of 16 edges --------------------------
__global__ __launch_bounds__(WPB * 32)
void pg_edge_mlp(const float* __restrict__ Z,
                 const int*   __restrict__ edge_index,
                 const float* __restrict__ W1,
                 const float* __restrict__ W2,
                 const float* __restrict__ b2p,
                 const float* __restrict__ W3,
                 const float* __restrict__ b3p,
                 const float* __restrict__ c1,
                 float* __restrict__ out,
                 int E, int ntiles) {
  __shared__ _Float16 hbuf[WPB][16 * LSTR];

  const int lane = threadIdx.x & 31;
  const int wave = threadIdx.x >> 5;
  const int c    = lane & 15;      // column / row-within-tile index
  const int hi   = lane >> 4;      // which half of the wave
  _Float16* lds  = hbuf[wave];

  const int koff16 = hi * 16;      // unused; B-fragments use koff8 as well
  const int koff8  = hi * 8;       // A-fragment K offset per lane half

  // ---- build register-resident B-fragments for W1 (2 parts x 2 ksteps x 4 ntiles)
  v16h wb1[2][2][4];
  #pragma unroll
  for (int p = 0; p < 2; ++p)
    #pragma unroll
    for (int ks = 0; ks < 2; ++ks)
      #pragma unroll
      for (int nt = 0; nt < 4; ++nt) {
        v16h f;
        const int ncol = nt * 16 + c;
        #pragma unroll
        for (int h = 0; h < 16; ++h) {
          const int k = p * 64 + ks * 32 + koff8 + (h & 7) + ((h >> 3) << 4);   // row of W1 block
          f[h] = (_Float16)W1[k * HH1 + ncol];
        }
        wb1[p][ks][nt] = f;
      }

  // ---- W2 B-fragments (2 ksteps x 2 ntiles), cols >= 20 zero-padded
  v16h wb2[2][2];
  #pragma unroll
  for (int ks = 0; ks < 2; ++ks)
    #pragma unroll
    for (int nt = 0; nt < 2; ++nt) {
      v16h f;
      const int ncol = nt * 16 + c;
      #pragma unroll
      for (int h = 0; h < 16; ++h) {
        const int k = ks * 32 + koff8 + (h & 7) + ((h >> 3) << 4);
        f[h] = (ncol < HH2) ? (_Float16)W2[k * HH2 + ncol] : (_Float16)0.0f;
      }
      wb2[ks][nt] = f;
    }

  // ---- per-lane constants (accumulator seeds, layer-3 weights)
  float c1v[4];
  #pragma unroll
  for (int nt = 0; nt < 4; ++nt) c1v[nt] = c1[nt * 16 + c];
  float b2v[2];
  #pragma unroll
  for (int nt = 0; nt < 2; ++nt) {
    const int ncol = nt * 16 + c;
    b2v[nt] = (ncol < HH2) ? b2p[ncol] : 0.0f;
  }
  float w3v0 = (c      < HH2) ? W3[c]      : 0.0f;
  float w3v1 = (16 + c < HH2) ? W3[16 + c] : 0.0f;
  const float b3 = b3p[0];

  const int gwave  = blockIdx.x * WPB + wave;
  const int nwaves = gridDim.x * WPB;

  // ---- software-pipelined edge-index loads (one tile ahead)
  int nri = 0, nrj = 0;
  if (gwave < ntiles) {
    int e0 = gwave * 16 + c;
    if (e0 >= E) e0 = E - 1;
    nri = edge_index[e0];
    nrj = edge_index[E + e0];
  }

  for (int tile = gwave; tile < ntiles; tile += nwaves) {
    const int ebase = tile * 16;
    const float* zi = Z + (size_t)nri * DD;
    const float* zj = Z + (size_t)nrj * DD;

    // issue next tile's index loads + row prefetch early; their latency
    // hides under this tile's WMMA work.
    const int tn = tile + nwaves;
    if (tn < ntiles) {
      int en = tn * 16 + c;
      if (en >= E) en = E - 1;
      nri = edge_index[en];
      nrj = edge_index[E + en];
      // each lane half prefetches one 128B line of its next rows
      __builtin_prefetch(Z + (size_t)nri * DD + hi * 32, 0, 3);
      __builtin_prefetch(Z + (size_t)nrj * DD + hi * 32, 0, 3);
    }

    // ---- layer 1: acc[nt] = relu( [z_i z_j] @ W1[:128] + c1 )
    v8f acc[4];
    #pragma unroll
    for (int nt = 0; nt < 4; ++nt) {
      v8f a;
      #pragma unroll
      for (int v = 0; v < 8; ++v) a[v] = c1v[nt];
      acc[nt] = a;
    }

    #pragma unroll
    for (int mtx = 0; mtx < 2; ++mtx) {            // z_i then z_j
      const float* zr = mtx ? zj : zi;
      #pragma unroll
      for (int ks = 0; ks < 2; ++ks) {             // K = 64 -> 2 x K32
        const int off = ks * 32 + koff8;
        // this lane's A-fragment slice: features off..off+7, off+16..off+23
        v4f a0 = *(const v4f*)(zr + off);
        v4f a1 = *(const v4f*)(zr + off + 4);
        v4f a2 = *(const v4f*)(zr + off + 16);
        v4f a3 = *(const v4f*)(zr + off + 20);
        v16h A;
        #pragma unroll
        for (int q = 0; q < 4; ++q) {
          A[q]      = (_Float16)a0[q];
          A[4 + q]  = (_Float16)a1[q];
          A[8 + q]  = (_Float16)a2[q];
          A[12 + q] = (_Float16)a3[q];
        }
        #pragma unroll
        for (int nt = 0; nt < 4; ++nt)
          acc[nt] = __builtin_amdgcn_wmma_f32_16x16x32_f16(
              false, A, false, wb1[mtx][ks][nt], (short)0, acc[nt],
              false, false);
      }
    }

    // ---- relu + stage h1 tile to LDS as f16, row-major [edge][feature]
    #pragma unroll
    for (int nt = 0; nt < 4; ++nt) {
      const int f = nt * 16 + c;                   // D-frag: col per lane
      #pragma unroll
      for (int v = 0; v < 8; ++v) {
        float x = acc[nt][v];
        x = x > 0.0f ? x : 0.0f;
        const int m = v + 8 * hi;                  // D-frag: row per VGPR
        lds[m * LSTR + f] = (_Float16)x;
      }
    }
    // per-wave private LDS region; same-wave DS ops are in order.

    // ---- layer 2: acc2[nt] = relu( h1 @ W2 + b2 ), K=64 -> 2 WMMA ksteps
    v8f acc2[2];
    #pragma unroll
    for (int nt = 0; nt < 2; ++nt) {
      v8f a;
      #pragma unroll
      for (int v = 0; v < 8; ++v) a[v] = b2v[nt];
      acc2[nt] = a;
    }
    #pragma unroll
    for (int ks = 0; ks < 2; ++ks) {
      const int base = ks * 32 + koff8;
      const int m = c;                             // A-frag: row per lane
      v8h lo = *(const v8h*)(lds + m * LSTR + base);
      v8h up = *(const v8h*)(lds + m * LSTR + base + 16);
      v16h A;
      #pragma unroll
      for (int q = 0; q < 8; ++q) { A[q] = lo[q]; A[8 + q] = up[q]; }
      #pragma unroll
      for (int nt = 0; nt < 2; ++nt)
        acc2[nt] = __builtin_amdgcn_wmma_f32_16x16x32_f16(
            false, A, false, wb2[ks][nt], (short)0, acc2[nt], false, false);
    }

    // ---- layer 3 (20x1) + sigmoid: per-lane partials, butterfly reduce
    float t[8];
    #pragma unroll
    for (int v = 0; v < 8; ++v) {
      float h0 = acc2[0][v]; h0 = h0 > 0.0f ? h0 : 0.0f;
      float h1 = acc2[1][v]; h1 = h1 > 0.0f ? h1 : 0.0f;
      t[v] = h0 * w3v0 + h1 * w3v1;                // cols >=20 are exact zeros
    }
    #pragma unroll
    for (int s = 1; s < 16; s <<= 1)
      #pragma unroll
      for (int v = 0; v < 8; ++v)
        t[v] += __shfl_xor(t[v], s, 32);

    if (c < 8) {                                   // lanes 0-7 / 16-23 store
      float om = t[0];
      #pragma unroll
      for (int v = 1; v < 8; ++v) om = (c == v) ? t[v] : om;
      const int eo = ebase + hi * 8 + c;
      if (eo < E)
        out[eo] = 1.0f / (1.0f + __expf(-(om + b3)));
    }
  }
}

extern "C" void kernel_launch(void* const* d_in, const int* in_sizes, int n_in,
                              void* d_out, int out_size, void* d_ws, size_t ws_size,
                              hipStream_t stream) {
  const float* Z          = (const float*)d_in[0];
  const int*   edge_index = (const int*)  d_in[1];
  const int*   node_idx   = (const int*)  d_in[2];
  const float* W1         = (const float*)d_in[3];
  const float* b1         = (const float*)d_in[4];
  const float* W2         = (const float*)d_in[5];
  const float* b2         = (const float*)d_in[6];
  const float* W3         = (const float*)d_in[7];
  const float* b3         = (const float*)d_in[8];
  float*       out        = (float*)d_out;

  const int E = in_sizes[1] / 2;
  float* c1 = (float*)d_ws;                        // 64 floats of scratch

  pg_precompute_c1<<<1, 64, 0, stream>>>(Z, node_idx, W1, b1, c1);

  const int ntiles = (E + 15) / 16;
  const int blocks = 512;                          // 4096 waves, grid-stride
  pg_edge_mlp<<<blocks, WPB * 32, 0, stream>>>(
      Z, edge_index, W1, W2, b2, W3, b3, c1, out, E, ntiles);
}
